// AOSPredictionLayer_68410239090891
// MI455X (gfx1250) — matrix operand order repaired and checked by reading, not compiled, not run on hardware
//
#include <hip/hip_runtime.h>

// ---------------- problem constants ----------------
#define RNUM 8
#define BDIM 512
#define NDIM 32
#define ROWS (BDIM * NDIM)        // 16384
#define DIN 512                   // concat input dim
#define H1D 512
#define H2D 256
#define H3D 128
#define MROWS 32                  // rows per workgroup (2 M-tiles)
#define MAXT (ROWS / MROWS)       // 512 row-tiles per relation (worst case)

// padded LDS strides (elements) — keep 16B alignment, break 64-bank stride
#define SX1 (512 + 8)
#define SX2 (256 + 8)
#define SH3 (128 + 4)

// dynamic LDS layout: region0 = X (later H2), region1 = H1 (later H3)
#define REG0_BYTES (MROWS * SX1 * 2)              // 33280
#define DYN_LDS    (2 * REG0_BYTES)               // 66560

typedef __attribute__((ext_vector_type(16))) __bf16 v16bf;
typedef __attribute__((ext_vector_type(8)))  float  v8f;
typedef __attribute__((ext_vector_type(8)))  unsigned short v8us;
typedef __attribute__((ext_vector_type(16))) unsigned short v16us;

__device__ __forceinline__ unsigned short f2bf(float f) {
  union { float f; unsigned int u; } c; c.f = f;
  unsigned int u = c.u;
  return (unsigned short)((u + 0x7FFFu + ((u >> 16) & 1u)) >> 16);
}

__device__ __forceinline__ v16bf make_frag(v8us lo, v8us hi) {
  v16us u;
#pragma unroll
  for (int i = 0; i < 8; ++i) { u[i] = lo[i]; u[8 + i] = hi[i]; }
  return __builtin_bit_cast(v16bf, u);
}

// One MLP layer on a (MT*16)-row tile: OUT = leaky(X @ W + bias)
// X: LDS bf16 row-major (stride SXs). W: global bf16 row-major [KDIM][NOUT].
// 8 waves; wave w owns N-tiles [w*TPW,(w+1)*TPW); each B-frag reused MT times.
template <int MT, int KDIM, int NOUT, int SXs, int SOs, bool F32OUT>
__device__ __forceinline__ void mlp_layer(const unsigned short* X,
                                          const unsigned short* W,
                                          const float* bias,
                                          unsigned short* OB, float* OF) {
  const int wave = threadIdx.x >> 5;
  const int lane = threadIdx.x & 31;
  constexpr int TPW = NOUT / 16 / 8;   // N-tiles per wave

  v8f acc[MT][TPW];
#pragma unroll
  for (int mt = 0; mt < MT; ++mt)
#pragma unroll
    for (int t = 0; t < TPW; ++t)
#pragma unroll
      for (int v = 0; v < 8; ++v) acc[mt][t][v] = 0.0f;

  const int m = lane & 15;
  const int half = lane >> 4;

  for (int kb = 0; kb < KDIM / 32; ++kb) {
    // A fragments (16x32 bf16) from LDS per ISA layout: two 16B chunks / lane
    v16bf a[MT];
#pragma unroll
    for (int mt = 0; mt < MT; ++mt) {
      const unsigned short* ap = X + (size_t)(mt * 16 + m) * SXs + kb * 32 + half * 8;
      a[mt] = make_frag(*(const v8us*)(ap), *(const v8us*)(ap + 16));
    }
    // B for this k-step: lane holds column n=(lane&15), k = half*8+i and 16+half*8+i
    const unsigned short* brow = W + (size_t)(kb * 32 + half * 8) * NOUT + m;
    __builtin_prefetch(brow + (size_t)32 * NOUT, 0, 0);  // next k-step
#pragma unroll
    for (int t = 0; t < TPW; ++t) {
      const unsigned short* bp = brow + (wave * TPW + t) * 16;
      v16us bu;
#pragma unroll
      for (int i = 0; i < 8; ++i) { bu[i] = bp[(size_t)i * NOUT]; bu[8 + i] = bp[(size_t)(16 + i) * NOUT]; }
      v16bf b = __builtin_bit_cast(v16bf, bu);
#pragma unroll
      for (int mt = 0; mt < MT; ++mt)
        acc[mt][t] = __builtin_amdgcn_wmma_f32_16x16x32_bf16(
            false, a[mt], false, b, (short)0, acc[mt][t], false, false);
    }
  }

  // epilogue: C layout -> lane column n = n0, rows mt*16 + (lane>>4)*8 + v
  const int n0 = lane & 15;
  const int mh = (lane >> 4) * 8;
#pragma unroll
  for (int mt = 0; mt < MT; ++mt)
#pragma unroll
    for (int t = 0; t < TPW; ++t) {
      const int n = (wave * TPW + t) * 16 + n0;
      const float bv = bias[n];
#pragma unroll
      for (int v = 0; v < 8; ++v) {
        float x = acc[mt][t][v] + bv;
        x = x > 0.0f ? x : 0.01f * x;               // LeakyReLU(0.01)
        if constexpr (F32OUT) OF[(size_t)(mt * 16 + mh + v) * SOs + n] = x;
        else                  OB[(size_t)(mt * 16 + mh + v) * SOs + n] = f2bf(x);
      }
    }
}

// ---------------- small utility kernels ----------------
__global__ void cvt_f32_bf16_kernel(const float* __restrict__ src,
                                    unsigned short* __restrict__ dst, int n) {
  int i = blockIdx.x * 256 + threadIdx.x;
  if (i < n) dst[i] = f2bf(src[i]);
}

__global__ void scatter_kernel(const int* __restrict__ s,
                               int* __restrict__ counts,
                               int* __restrict__ buckets) {
  int g = blockIdx.x * 256 + threadIdx.x;
  if (g < ROWS) {
    int r = s[g];
    int pos = atomicAdd(counts + r, 1);
    buckets[(r << 14) + pos] = g;
  }
}

// ---------------- UI MLP: [512,512] -> 512 -> 256 -> 128 ----------------
__launch_bounds__(256)
__global__ void ui_kernel(const float* __restrict__ u_emb, const float* __restrict__ i_emb,
                          const unsigned short* __restrict__ Wu1b,
                          const unsigned short* __restrict__ Wu2b,
                          const unsigned short* __restrict__ Wu3b,
                          const float* __restrict__ bu1, const float* __restrict__ bu2,
                          const float* __restrict__ bu3, float* __restrict__ ui_out) {
  const int tile = blockIdx.x;  // 0..15, 32 rows each
  extern __shared__ __attribute__((aligned(16))) char smem[];
  unsigned short* sX  = (unsigned short*)smem;                // [32][SX1]
  unsigned short* sH1 = (unsigned short*)(smem + REG0_BYTES); // [32][SX1]
  unsigned short* sH2 = (unsigned short*)smem;                // reuse X region
  float*          sH3 = (float*)(smem + REG0_BYTES);          // reuse H1 region

  // stage concat(u,i) rows as bf16 (float4 loads)
  for (int e = threadIdx.x; e < MROWS * 128; e += 256) {
    int mm = e >> 7, c4 = (e & 127) << 2;
    int row = tile * MROWS + mm;
    const float* src = (c4 < 256) ? (u_emb + (size_t)row * 256 + c4)
                                  : (i_emb + (size_t)row * 256 + (c4 - 256));
    float4 f = *(const float4*)src;
    unsigned short* dst = sX + (size_t)mm * SX1 + c4;
    dst[0] = f2bf(f.x); dst[1] = f2bf(f.y); dst[2] = f2bf(f.z); dst[3] = f2bf(f.w);
  }
  __syncthreads();
  mlp_layer<2, DIN, H1D, SX1, SX1, false>(sX, Wu1b, bu1, sH1, nullptr);
  __syncthreads();
  mlp_layer<2, H1D, H2D, SX1, SX2, false>(sH1, Wu2b, bu2, sH2, nullptr);
  __syncthreads();
  mlp_layer<2, H2D, H3D, SX2, SH3, true>(sH2, Wu3b, bu3, nullptr, sH3);
  __syncthreads();
  for (int e = threadIdx.x; e < MROWS * 128; e += 256) {
    int mm = e >> 7, c = e & 127;
    ui_out[(size_t)(tile * MROWS + mm) * 128 + c] = sH3[(size_t)mm * SH3 + c];
  }
}

// ------- AO routed MLP + dot: 32 gathered rows of one relation per WG -------
__launch_bounds__(256)
__global__ void ao_kernel(const float* __restrict__ a_emb, const float* __restrict__ o_emb,
                          const unsigned short* __restrict__ W1b,
                          const unsigned short* __restrict__ W2b,
                          const unsigned short* __restrict__ W3b,
                          const float* __restrict__ b1, const float* __restrict__ b2,
                          const float* __restrict__ b3,
                          const int* __restrict__ counts, const int* __restrict__ buckets,
                          const float* __restrict__ ui_out, float* __restrict__ pred) {
  const int rel  = blockIdx.x >> 9;         // / MAXT
  const int tile = blockIdx.x & (MAXT - 1);
  const int cnt  = counts[rel];
  if (tile * MROWS >= cnt) return;          // uniform exit: EXEC stays all-1s

  extern __shared__ __attribute__((aligned(16))) char smem[];
  unsigned short* sX  = (unsigned short*)smem;                // [32][SX1]
  unsigned short* sH1 = (unsigned short*)(smem + REG0_BYTES); // [32][SX1]
  unsigned short* sH2 = (unsigned short*)smem;                // reuse X region
  float*          sH3 = (float*)(smem + REG0_BYTES);          // reuse H1 region

  __shared__ int   sRow[MROWS];   // -1 when padded
  __shared__ int   sRowC[MROWS];  // clamped (always valid for compute)
  __shared__ float sP[MROWS][8];

  if (threadIdx.x < MROWS) {
    int ridx = tile * MROWS + (int)threadIdx.x;
    int g0 = buckets[(rel << 14) + tile * MROWS];
    int g  = (ridx < cnt) ? buckets[(rel << 14) + ridx] : g0;
    sRowC[threadIdx.x] = g;
    sRow[threadIdx.x]  = (ridx < cnt) ? g : -1;
  }
  __syncthreads();

  // stage concat(a,o) gathered rows as bf16
  for (int e = threadIdx.x; e < MROWS * 128; e += 256) {
    int mm = e >> 7, c4 = (e & 127) << 2;
    int g = sRowC[mm];
    const float* src = (c4 < 256) ? (a_emb + (size_t)g * 256 + c4)
                                  : (o_emb + (size_t)g * 256 + (c4 - 256));
    float4 f = *(const float4*)src;
    unsigned short* dst = sX + (size_t)mm * SX1 + c4;
    dst[0] = f2bf(f.x); dst[1] = f2bf(f.y); dst[2] = f2bf(f.z); dst[3] = f2bf(f.w);
  }
  __syncthreads();

  mlp_layer<2, DIN, H1D, SX1, SX1, false>(sX,
      W1b + (size_t)rel * DIN * H1D, b1 + rel * H1D, sH1, nullptr);
  __syncthreads();
  mlp_layer<2, H1D, H2D, SX1, SX2, false>(sH1,
      W2b + (size_t)rel * H1D * H2D, b2 + rel * H2D, sH2, nullptr);
  __syncthreads();
  mlp_layer<2, H2D, H3D, SX2, SH3, true>(sH2,
      W3b + (size_t)rel * H2D * H3D, b3 + rel * H3D, nullptr, sH3);
  __syncthreads();

  // pred[row] = dot(ui_emb[b(row)], h3[row]) over 128; 32 rows x 8 segments
  {
    int mm = threadIdx.x >> 3, seg = threadIdx.x & 7;
    int g = sRowC[mm];
    int b = g >> 5;  // row -> batch (N=32)
    const float* u = ui_out + (size_t)b * 128 + seg * 16;
    float p = 0.0f;
#pragma unroll
    for (int j = 0; j < 16; ++j) p += sH3[(size_t)mm * SH3 + seg * 16 + j] * u[j];
    sP[mm][seg] = p;
  }
  __syncthreads();
  if (threadIdx.x < MROWS) {
    int g = sRow[threadIdx.x];
    if (g >= 0) {
      float p = 0.0f;
#pragma unroll
      for (int j = 0; j < 8; ++j) p += sP[threadIdx.x][j];
      pred[g] = p;
    }
  }
}

// ---------------- launcher ----------------
extern "C" void kernel_launch(void* const* d_in, const int* in_sizes, int n_in,
                              void* d_out, int out_size, void* d_ws, size_t ws_size,
                              hipStream_t stream) {
  (void)in_sizes; (void)n_in; (void)out_size; (void)ws_size;
  const float* u_emb = (const float*)d_in[0];
  const float* i_emb = (const float*)d_in[1];
  const float* a_emb = (const float*)d_in[2];
  const float* o_emb = (const float*)d_in[3];
  const int*   s     = (const int*)d_in[4];
  const float* W1 = (const float*)d_in[5];   const float* b1 = (const float*)d_in[6];
  const float* W2 = (const float*)d_in[7];   const float* b2 = (const float*)d_in[8];
  const float* W3 = (const float*)d_in[9];   const float* b3 = (const float*)d_in[10];
  const float* Wu1 = (const float*)d_in[11]; const float* bu1 = (const float*)d_in[12];
  const float* Wu2 = (const float*)d_in[13]; const float* bu2 = (const float*)d_in[14];
  const float* Wu3 = (const float*)d_in[15]; const float* bu3 = (const float*)d_in[16];
  float* pred = (float*)d_out;

  char* w = (char*)d_ws;
  int* counts = (int*)w;                      w += 256;
  int* buckets = (int*)w;                     w += (size_t)RNUM * ROWS * 4;
  float* ui_out = (float*)w;                  w += (size_t)BDIM * H3D * 4;
  unsigned short* W1b = (unsigned short*)w;   w += (size_t)RNUM * DIN * H1D * 2;
  unsigned short* W2b = (unsigned short*)w;   w += (size_t)RNUM * H1D * H2D * 2;
  unsigned short* W3b = (unsigned short*)w;   w += (size_t)RNUM * H2D * H3D * 2;
  unsigned short* Wu1b = (unsigned short*)w;  w += (size_t)DIN * H1D * 2;
  unsigned short* Wu2b = (unsigned short*)w;  w += (size_t)H1D * H2D * 2;
  unsigned short* Wu3b = (unsigned short*)w;  w += (size_t)H2D * H3D * 2;

  hipMemsetAsync(counts, 0, RNUM * sizeof(int), stream);

  auto cvt = [&](const float* srcp, unsigned short* dstp, int n) {
    cvt_f32_bf16_kernel<<<(n + 255) / 256, 256, 0, stream>>>(srcp, dstp, n);
  };
  cvt(W1, W1b, RNUM * DIN * H1D);
  cvt(W2, W2b, RNUM * H1D * H2D);
  cvt(W3, W3b, RNUM * H2D * H3D);
  cvt(Wu1, Wu1b, DIN * H1D);
  cvt(Wu2, Wu2b, H1D * H2D);
  cvt(Wu3, Wu3b, H2D * H3D);

  scatter_kernel<<<ROWS / 256, 256, 0, stream>>>(s, counts, buckets);

  ui_kernel<<<BDIM / MROWS, 256, DYN_LDS, stream>>>(u_emb, i_emb, Wu1b, Wu2b, Wu3b,
                                                    bu1, bu2, bu3, ui_out);

  ao_kernel<<<RNUM * MAXT, 256, DYN_LDS, stream>>>(a_emb, o_emb, W1b, W2b, W3b,
                                                   b1, b2, b3, counts, buckets,
                                                   ui_out, pred);
}
